// MLP_89498528514757
// MI455X (gfx1250) — compile-verified, matrix operand order checked by reading, not run on hardware
//
#include <hip/hip_runtime.h>

typedef _Float16 f16;
typedef __attribute__((ext_vector_type(16))) _Float16 v16h;
typedef __attribute__((ext_vector_type(8)))  _Float16 v8h;
typedef __attribute__((ext_vector_type(8)))  float    v8f;

#define NUSERS 100000
#define BATCH  1024
#define D1 256   // layer-1 width (gather output)
#define D2 128   // layer-2 width
#define D3 64    // layer-3 width

// LDS row pitches padded by 8 f16 (16B) so that lane i reading row i hits
// rotating bank groups (row stride = 528B / 272B / 144B -> 4-dword bank skew).
#define LD1 (D1 + 8)
#define LD2 (D2 + 8)
#define LD3 (D3 + 8)

// ---------------------------------------------------------------------------
// WMMA f16 fragment loader for v_wmma_f32_16x16x32_f16.
// Source is row-major [16 rows][ld] (A: rows=M, B pre-transposed: rows=N).
// ISA 16-bit A/B layout: lanes 0-15 hold index L with K {kb+0..7, kb+16..23},
// lanes 16-31 hold index L-16 with K {kb+8..15, kb+24..31}.
// Each half is one contiguous 16-byte chunk -> ds_load_b128 / global b128.
// ---------------------------------------------------------------------------
__device__ __forceinline__ v16h load_frag(const f16* base, int ld, int kbase,
                                          int lane) {
  const int half = lane >> 4;      // 0 or 1
  const int idx  = lane & 15;      // row (A) / col (B) within the 16-tile
  const f16* p = base + idx * ld + kbase + half * 8;
  v8h c0 = *(const v8h*)(p);        // K = kbase + half*8 + 0..7
  v8h c1 = *(const v8h*)(p + 16);   // K = kbase + 16 + half*8 + 0..7
  return __builtin_shufflevector(c0, c1, 0, 1, 2, 3, 4, 5, 6, 7,
                                 8, 9, 10, 11, 12, 13, 14, 15);
}

// ---------------------------------------------------------------------------
// Prep: W [K][N] f32 row-major  ->  Wt [N][K] f16 row-major.
// Transposed-f16 weights make every B-fragment lane load a contiguous 16B run.
// Tiny (80 KB f16 total), runs once per launch before the fused kernel.
// ---------------------------------------------------------------------------
__global__ void mlp_transpose_f32_to_f16(const float* __restrict__ src,
                                         f16* __restrict__ dst, int K, int N) {
  int i = blockIdx.x * blockDim.x + threadIdx.x;
  if (i < K * N) {
    int k = i / N;
    int n = i - k * N;
    dst[n * K + k] = (f16)src[i];
  }
}

// ---------------------------------------------------------------------------
// Fused MLP: gather+L1 -> LDS, L2 (WMMA, K=256), L3 (WMMA, K=128), L4 (dot).
// One block = 16 batch rows, 256 threads = 8 wave32s.
// WMMA layers are software-pipelined: ALL K-step fragments are issued first
// (in-order completion per counter type), then the WMMA chain runs so the
// compiler can use partial s_wait_loadcnt/dscnt values instead of a full
// drain before every matrix op.
// ---------------------------------------------------------------------------
__global__ void __launch_bounds__(256)
mlp_fused_kernel(const int* __restrict__ user_ids,
                 const int* __restrict__ item_ids,
                 const float* __restrict__ W1, const float* __restrict__ b1,
                 const f16* __restrict__ w2t,  const float* __restrict__ b2,
                 const f16* __restrict__ w3t,  const float* __restrict__ b3,
                 const float* __restrict__ W4, const float* __restrict__ b4,
                 float* __restrict__ out) {
  __shared__ __align__(16) f16 lds_h1[16 * LD1];  // 8448 B
  __shared__ __align__(16) f16 lds_h2[16 * LD2];  // 4352 B
  __shared__ __align__(16) f16 lds_h3[16 * LD3];  // 2304 B

  const int tid   = threadIdx.x;
  const int lane  = tid & 31;
  const int wave  = tid >> 5;
  const int half  = lane >> 4;
  const int idx   = lane & 15;
  const int mrow0 = blockIdx.x * 16;

  // ---- Layer 1: two-row gather + bias + ReLU, f32 -> f16 into LDS --------
  // thread = column (coalesced over W1 rows, which live hot in the 192MB L2).
  {
    const int c = tid;  // 0..255 == D1
    const float bias = b1[c];
#pragma unroll 4
    for (int rr = 0; rr < 16; ++rr) {
      const int row = mrow0 + rr;
      const int u  = user_ids[row];
      const int it = item_ids[row];
      float v = W1[(size_t)u * D1 + c] + W1[(size_t)(NUSERS + it) * D1 + c] + bias;
      v = v > 0.0f ? v : 0.0f;
      lds_h1[rr * LD1 + c] = (f16)v;
    }
  }
  __syncthreads();

  // ---- Layer 2: [16,256] @ [256,128] via WMMA; wave w owns N-tile w ------
  {
    const int n0 = wave * 16;               // all 8 waves active
    const f16* bbase = w2t + (size_t)n0 * D1;

    // Preload all 8 K-steps (8 ds_load_b128 pairs + 8 global_load_b128 pairs)
    v16h afrag[8], bfrag[8];
#pragma unroll
    for (int kk = 0; kk < 8; ++kk) {
      afrag[kk] = load_frag(lds_h1, LD1, kk * 32, lane);
      bfrag[kk] = load_frag(bbase, D1, kk * 32, lane);
    }
    // WMMA chain: in-order load completion lets waits be partial.
    v8f acc = {};
#pragma unroll
    for (int kk = 0; kk < 8; ++kk) {
      acc = __builtin_amdgcn_wmma_f32_16x16x32_f16(
          false, afrag[kk], false, bfrag[kk], (short)0, acc, false, false);
    }

    // D layout: lane<16 -> M=r,   N=lane;  lane>=16 -> M=8+r, N=lane-16.
    const int n = n0 + idx;
    const float bias = b2[n];
#pragma unroll
    for (int r = 0; r < 8; ++r) {
      const int m = half * 8 + r;
      float v = acc[r] + bias;
      v = v > 0.0f ? v : 0.0f;
      lds_h2[m * LD2 + n] = (f16)v;
    }
  }
  __syncthreads();

  // ---- Layer 3: [16,128] @ [128,64] via WMMA; waves 0..3 active ----------
  if (wave < 4) {                            // wave-uniform branch: EXEC all-1s
    const int n0 = wave * 16;
    const f16* bbase = w3t + (size_t)n0 * D2;

    v16h afrag[4], bfrag[4];
#pragma unroll
    for (int kk = 0; kk < 4; ++kk) {
      afrag[kk] = load_frag(lds_h2, LD2, kk * 32, lane);
      bfrag[kk] = load_frag(bbase, D2, kk * 32, lane);
    }
    v8f acc = {};
#pragma unroll
    for (int kk = 0; kk < 4; ++kk) {
      acc = __builtin_amdgcn_wmma_f32_16x16x32_f16(
          false, afrag[kk], false, bfrag[kk], (short)0, acc, false, false);
    }

    const int n = n0 + idx;
    const float bias = b3[n];
#pragma unroll
    for (int r = 0; r < 8; ++r) {
      const int m = half * 8 + r;
      float v = acc[r] + bias;
      v = v > 0.0f ? v : 0.0f;
      lds_h3[m * LD3 + n] = (f16)v;
    }
  }
  __syncthreads();

  // ---- Layer 4: [16,64] @ [64,1] -- too thin for the matrix pipe ---------
  if (tid < 16) {
    const int m = tid;
    float s = b4[0];
#pragma unroll
    for (int k = 0; k < D3; ++k)
      s += (float)lds_h3[m * LD3 + k] * W4[k];
    out[mrow0 + m] = s;
  }
}

// ---------------------------------------------------------------------------
extern "C" void kernel_launch(void* const* d_in, const int* in_sizes, int n_in,
                              void* d_out, int out_size, void* d_ws,
                              size_t ws_size, hipStream_t stream) {
  const int*   user_ids = (const int*)d_in[0];
  const int*   item_ids = (const int*)d_in[1];
  const float* W1 = (const float*)d_in[2];
  const float* b1 = (const float*)d_in[3];
  const float* W2 = (const float*)d_in[4];
  const float* b2 = (const float*)d_in[5];
  const float* W3 = (const float*)d_in[6];
  const float* b3 = (const float*)d_in[7];
  const float* W4 = (const float*)d_in[8];
  const float* b4 = (const float*)d_in[9];
  float* out = (float*)d_out;

  // Workspace: W2^T f16 (128x256 = 64KB), then W3^T f16 (64x128 = 16KB).
  f16* w2t = (f16*)d_ws;
  f16* w3t = (f16*)((char*)d_ws + (size_t)D1 * D2 * sizeof(f16));

  // Prep: transpose+downconvert weights (ordered on `stream`).
  mlp_transpose_f32_to_f16<<<(D1 * D2 + 255) / 256, 256, 0, stream>>>(
      W2, w2t, D1, D2);
  mlp_transpose_f32_to_f16<<<(D2 * D3 + 255) / 256, 256, 0, stream>>>(
      W3, w3t, D2, D3);

  // Fused MLP: 1024 rows / 16 rows per block = 64 blocks of 8 wave32s.
  mlp_fused_kernel<<<BATCH / 16, 256, 0, stream>>>(
      user_ids, item_ids, W1, b1, w2t, b2, w3t, b3, W4, b4, out);
}
